// GptOssAttention_17824114279001
// MI455X (gfx1250) — compile-verified, matrix operand order checked by reading, not run on hardware
//
#include <hip/hip_runtime.h>
#include <hip/hip_bf16.h>
#include <math.h>

typedef __attribute__((ext_vector_type(16))) __bf16 v16bf;
typedef __attribute__((ext_vector_type(8)))  __bf16 v8bf;
typedef __attribute__((ext_vector_type(8)))  float  v8f;
typedef __attribute__((ext_vector_type(4)))  int    v4i;

#if defined(__gfx1250__) && __has_builtin(__builtin_amdgcn_global_load_async_to_lds_b128)
#define USE_ASYNC 1
#else
#define USE_ASYNC 0
#endif

namespace {
constexpr int kS    = 1024;
constexpr int kH    = 2880;
constexpr int kNH   = 64;
constexpr int kNKV  = 8;
constexpr int kHD   = 64;
constexpr int kQKV  = (kNH + 2 * kNKV) * kHD;  // 5120
constexpr int kQD   = kNH * kHD;               // 4096
constexpr int kKVD  = kNKV * kHD;              // 512
constexpr float kScale    = 0.125f;              // 64^-0.5
constexpr float kLogTheta = 9.2103403719761836f; // ln(10000)
}

__device__ __forceinline__ v8f wmma_bf16(v16bf a, v16bf b, v8f c) {
  return __builtin_amdgcn_wmma_f32_16x16x32_bf16(
      /*neg_a=*/false, a, /*neg_b=*/false, b,
      /*c_mod=*/(short)0, c, /*reuse_a=*/false, /*reuse_b=*/false);
}

// Async copy of 16 bytes global -> LDS (ASYNCcnt path); sync fallback.
__device__ __forceinline__ void cp_b128(__bf16* lds, const __bf16* g) {
#if USE_ASYNC
  __builtin_amdgcn_global_load_async_to_lds_b128(
      (__attribute__((address_space(1))) v4i*)g,
      (__attribute__((address_space(3))) v4i*)lds, 0, 0);
#else
  *(v8bf*)lds = *(const v8bf*)g;
#endif
}

__device__ __forceinline__ void wait_async0() {
#if USE_ASYNC
#if __has_builtin(__builtin_amdgcn_s_wait_asynccnt)
  __builtin_amdgcn_s_wait_asynccnt(0);
#else
  asm volatile("s_wait_asynccnt 0x0" ::: "memory");
#endif
#endif
}

// Load one 16-bit A/B fragment for a 32-wide K chunk.
// ISA layout: lanes 0-15 hold K {0..7, 16..23}; lanes 16-31 hold K {8..15, 24..31}.
__device__ __forceinline__ v16bf ld_frag32(const __bf16* base, int hf) {
  const v8bf lo = *(const v8bf*)(base + hf * 8);
  const v8bf hi = *(const v8bf*)(base + 16 + hf * 8);
  return __builtin_shufflevector(lo, hi, 0, 1, 2, 3, 4, 5, 6, 7,
                                 8, 9, 10, 11, 12, 13, 14, 15);
}

// ---------------------------------------------------------------- f32 -> bf16
__global__ void cvt_kernel(const float* __restrict__ src,
                           __bf16* __restrict__ dst, int n) {
  int i = (blockIdx.x * blockDim.x + threadIdx.x) * 8;
  if (i < n) {
    float4 a = *(const float4*)(src + i);
    float4 b = *(const float4*)(src + i + 4);
    v8bf o = {(__bf16)a.x, (__bf16)a.y, (__bf16)a.z, (__bf16)a.w,
              (__bf16)b.x, (__bf16)b.y, (__bf16)b.z, (__bf16)b.w};
    *(v8bf*)(dst + i) = o;
  }
}

// ------------------------------------------------- QKV GEMM + fused RoPE
// C[s, n] = sum_k hs[s,k] * qkv_w[n,k];  M=1024, N=5120, K=2880.
// Block tile 128(M)x256(N), 8 waves, each wave 64x64 (16 WMMAs / k-step).
// Double-buffered LDS, async global->LDS copies overlap WMMA.
__global__ __launch_bounds__(256) void qkv_gemm_rope_kernel(
    const __bf16* __restrict__ A, const __bf16* __restrict__ W,
    const int* __restrict__ pos,
    __bf16* __restrict__ qb, __bf16* __restrict__ kbuf,
    __bf16* __restrict__ vtb) {
  __shared__ __bf16 As[2][128][40];
  __shared__ __bf16 Bs[2][256][40];
  const int tid  = threadIdx.x;
  const int lane = tid & 31;
  const int wave = tid >> 5;
  const int wm   = wave >> 2;        // 0..1 (M, 64 rows each)
  const int wn   = wave & 3;         // 0..3 (N, 64 cols each)
  const int ln   = lane & 15;
  const int hf   = lane >> 4;
  const int m0   = blockIdx.y * 128;
  const int n0   = blockIdx.x * 256;
  const int lrow = tid >> 1;         // 0..127
  const int lseg = (tid & 1) * 16;   // 0 or 16

  v8f acc[4][4] = {};
  const __bf16* aptr  = A + (size_t)(m0 + lrow) * kH + lseg;
  const __bf16* bptr0 = W + (size_t)(n0 + lrow) * kH + lseg;
  const __bf16* bptr1 = W + (size_t)(n0 + 128 + lrow) * kH + lseg;

  constexpr int ntile = kH / 32;  // 90
  // Prologue: stage tile 0 into buffer 0.
  cp_b128(&As[0][lrow][lseg],       aptr);
  cp_b128(&As[0][lrow][lseg + 8],   aptr + 8);
  cp_b128(&Bs[0][lrow][lseg],       bptr0);
  cp_b128(&Bs[0][lrow][lseg + 8],   bptr0 + 8);
  cp_b128(&Bs[0][lrow + 128][lseg],     bptr1);
  cp_b128(&Bs[0][lrow + 128][lseg + 8], bptr1 + 8);

  for (int t = 0; t < ntile; ++t) {
    const int cur = t & 1;
    wait_async0();
    __syncthreads();
    if (t + 1 < ntile) {
      const int k0 = (t + 1) * 32;
      const int nxt = cur ^ 1;
      cp_b128(&As[nxt][lrow][lseg],       aptr + k0);
      cp_b128(&As[nxt][lrow][lseg + 8],   aptr + k0 + 8);
      cp_b128(&Bs[nxt][lrow][lseg],       bptr0 + k0);
      cp_b128(&Bs[nxt][lrow][lseg + 8],   bptr0 + k0 + 8);
      cp_b128(&Bs[nxt][lrow + 128][lseg],     bptr1 + k0);
      cp_b128(&Bs[nxt][lrow + 128][lseg + 8], bptr1 + k0 + 8);
    }
    v16bf af[4], bf[4];
#pragma unroll
    for (int mt = 0; mt < 4; ++mt)
      af[mt] = ld_frag32(&As[cur][wm * 64 + mt * 16 + ln][0], hf);
#pragma unroll
    for (int nt = 0; nt < 4; ++nt)
      bf[nt] = ld_frag32(&Bs[cur][wn * 64 + nt * 16 + ln][0], hf);
#pragma unroll
    for (int mt = 0; mt < 4; ++mt)
#pragma unroll
      for (int nt = 0; nt < 4; ++nt)
        acc[mt][nt] = wmma_bf16(af[mt], bf[nt], acc[mt][nt]);
  }

  const int gn0   = n0 + wn * 64;   // head-aligned (64-wide)
  const int mbase = m0 + wm * 64;

  // Fused RoPE for Q and K regions: pair (d, d+32) lives at (nt, nt+2).
  if (gn0 < kQD + kKVD) {
#pragma unroll
    for (int mt = 0; mt < 4; ++mt) {
#pragma unroll
      for (int nt = 0; nt < 2; ++nt) {
        const int d = nt * 16 + ln;  // 0..31
        const float inv = __expf((float)d * (-kLogTheta / 32.f));
#pragma unroll
        for (int r = 0; r < 8; ++r) {
          const int s = mbase + mt * 16 + hf * 8 + r;
          const float ang = (float)pos[s] * inv;
          float sn, cs;
          __sincosf(ang, &sn, &cs);
          const float x1 = acc[mt][nt][r];
          const float x2 = acc[mt][nt + 2][r];
          acc[mt][nt][r]     = x1 * cs - x2 * sn;
          acc[mt][nt + 2][r] = x2 * cs + x1 * sn;
        }
      }
    }
  }

#pragma unroll
  for (int mt = 0; mt < 4; ++mt)
#pragma unroll
    for (int nt = 0; nt < 4; ++nt)
#pragma unroll
      for (int r = 0; r < 8; ++r) {
        const int n = gn0 + nt * 16 + ln;
        const int s = mbase + mt * 16 + hf * 8 + r;
        const __bf16 v = (__bf16)acc[mt][nt][r];
        if (n < kQD) {
          qb[(size_t)s * kQD + n] = v;                 // [S][NH*HD]
        } else if (n < kQD + kKVD) {
          kbuf[(size_t)s * kKVD + (n - kQD)] = v;      // [S][NKV*HD]
        } else {
          const int nv = n - (kQD + kKVD);             // g*64 + d
          vtb[(size_t)nv * kS + s] = v;                // V^T: [g][d][S]
        }
      }
}

// --------------------------------------------- Flash attention with sinks
// One wave per (head h, 16-query tile). Online softmax init: m=sink, l=1.
__global__ __launch_bounds__(32) void attn_kernel(
    const __bf16* __restrict__ qb, const __bf16* __restrict__ kbuf,
    const __bf16* __restrict__ vtb, const float* __restrict__ sinks,
    __bf16* __restrict__ ob) {
  __shared__ __bf16 Ps[16][32];
  const int h    = blockIdx.x >> 6;  // 0..63
  const int qt   = blockIdx.x & 63;  // q tile (16 rows)
  const int g    = h >> 3;           // kv head
  const int lane = threadIdx.x;
  const int ln   = lane & 15;
  const int hf   = lane >> 4;

  v16bf qf[2];
  {
    const __bf16* qptr = qb + (size_t)(qt * 16 + ln) * kQD + h * kHD;
    qf[0] = ld_frag32(qptr, hf);
    qf[1] = ld_frag32(qptr + 32, hf);
  }
  const float snk = sinks[h];
  float m_run[8], l_run[8];
#pragma unroll
  for (int r = 0; r < 8; ++r) { m_run[r] = snk; l_run[r] = 1.f; }
  v8f oacc[4] = {};

  const int nchunk = (qt * 16 + 15) / 32 + 1;  // causal: only needed chunks
  for (int kc = 0; kc < nchunk; ++kc) {
    const int kb0 = kc * 32;
    v8f lt[2] = {};
#pragma unroll
    for (int j = 0; j < 2; ++j) {
      const __bf16* kptr = kbuf + (size_t)(kb0 + j * 16 + ln) * kKVD + g * kHD;
      lt[j] = wmma_bf16(qf[0], ld_frag32(kptr, hf), lt[j]);
      lt[j] = wmma_bf16(qf[1], ld_frag32(kptr + 32, hf), lt[j]);
    }
    float alpha[8];
#pragma unroll
    for (int r = 0; r < 8; ++r) {
      const int qrow = qt * 16 + hf * 8 + r;
#pragma unroll
      for (int j = 0; j < 2; ++j) {
        const int key = kb0 + j * 16 + ln;
        lt[j][r] = (key <= qrow) ? lt[j][r] * kScale : -__builtin_inff();
      }
      float t = fmaxf(lt[0][r], lt[1][r]);
#pragma unroll
      for (int msk = 1; msk < 16; msk <<= 1)
        t = fmaxf(t, __shfl_xor(t, msk, 32));
      const float mnew = fmaxf(m_run[r], t);
      const float p0 = __expf(lt[0][r] - mnew);
      const float p1 = __expf(lt[1][r] - mnew);
      lt[0][r] = p0;
      lt[1][r] = p1;
      float rs = p0 + p1;
#pragma unroll
      for (int msk = 1; msk < 16; msk <<= 1) rs += __shfl_xor(rs, msk, 32);
      alpha[r] = __expf(m_run[r] - mnew);
      l_run[r] = l_run[r] * alpha[r] + rs;
      m_run[r] = mnew;
    }
#pragma unroll
    for (int dt = 0; dt < 4; ++dt)
#pragma unroll
      for (int r = 0; r < 8; ++r) oacc[dt][r] *= alpha[r];

    // C-layout P -> A-layout via LDS bounce.
    __syncthreads();
#pragma unroll
    for (int j = 0; j < 2; ++j)
#pragma unroll
      for (int r = 0; r < 8; ++r)
        Ps[hf * 8 + r][j * 16 + ln] = (__bf16)lt[j][r];
    __syncthreads();
    const v16bf pf = ld_frag32(&Ps[ln][0], hf);
#pragma unroll
    for (int dt = 0; dt < 4; ++dt) {
      const __bf16* vptr = vtb + (size_t)(g * kHD + dt * 16 + ln) * kS + kb0;
      oacc[dt] = wmma_bf16(pf, ld_frag32(vptr, hf), oacc[dt]);
    }
  }
#pragma unroll
  for (int dt = 0; dt < 4; ++dt)
#pragma unroll
    for (int r = 0; r < 8; ++r) {
      const int s = qt * 16 + hf * 8 + r;
      ob[(size_t)s * kQD + h * kHD + dt * 16 + ln] =
          (__bf16)(oacc[dt][r] / l_run[r]);
    }
}

// ----------------------------------------------------------- O projection
// out[s, n] = sum_k attn[s,k] * o_w[n,k];  M=1024, N=2880, K=4096.
__global__ __launch_bounds__(256) void o_gemm_kernel(
    const __bf16* __restrict__ A, const __bf16* __restrict__ W,
    float* __restrict__ out) {
  __shared__ __bf16 As[2][128][40];
  __shared__ __bf16 Bs[2][256][40];
  const int tid  = threadIdx.x;
  const int lane = tid & 31;
  const int wave = tid >> 5;
  const int wm   = wave >> 2;
  const int wn   = wave & 3;
  const int ln   = lane & 15;
  const int hf   = lane >> 4;
  const int m0   = blockIdx.y * 128;
  const int n0   = blockIdx.x * 256;
  const int lrow = tid >> 1;
  const int lseg = (tid & 1) * 16;

  v8f acc[4][4] = {};
  const int nrow0 = (n0 + lrow < kH) ? (n0 + lrow) : (kH - 1);
  const int nrow1 = (n0 + 128 + lrow < kH) ? (n0 + 128 + lrow) : (kH - 1);
  const __bf16* aptr  = A + (size_t)(m0 + lrow) * kQD + lseg;
  const __bf16* bptr0 = W + (size_t)nrow0 * kQD + lseg;
  const __bf16* bptr1 = W + (size_t)nrow1 * kQD + lseg;

  constexpr int ntile = kQD / 32;  // 128
  cp_b128(&As[0][lrow][lseg],       aptr);
  cp_b128(&As[0][lrow][lseg + 8],   aptr + 8);
  cp_b128(&Bs[0][lrow][lseg],       bptr0);
  cp_b128(&Bs[0][lrow][lseg + 8],   bptr0 + 8);
  cp_b128(&Bs[0][lrow + 128][lseg],     bptr1);
  cp_b128(&Bs[0][lrow + 128][lseg + 8], bptr1 + 8);

  for (int t = 0; t < ntile; ++t) {
    const int cur = t & 1;
    wait_async0();
    __syncthreads();
    if (t + 1 < ntile) {
      const int k0 = (t + 1) * 32;
      const int nxt = cur ^ 1;
      cp_b128(&As[nxt][lrow][lseg],       aptr + k0);
      cp_b128(&As[nxt][lrow][lseg + 8],   aptr + k0 + 8);
      cp_b128(&Bs[nxt][lrow][lseg],       bptr0 + k0);
      cp_b128(&Bs[nxt][lrow][lseg + 8],   bptr0 + k0 + 8);
      cp_b128(&Bs[nxt][lrow + 128][lseg],     bptr1 + k0);
      cp_b128(&Bs[nxt][lrow + 128][lseg + 8], bptr1 + k0 + 8);
    }
    v16bf af[4], bf[4];
#pragma unroll
    for (int mt = 0; mt < 4; ++mt)
      af[mt] = ld_frag32(&As[cur][wm * 64 + mt * 16 + ln][0], hf);
#pragma unroll
    for (int nt = 0; nt < 4; ++nt)
      bf[nt] = ld_frag32(&Bs[cur][wn * 64 + nt * 16 + ln][0], hf);
#pragma unroll
    for (int mt = 0; mt < 4; ++mt)
#pragma unroll
      for (int nt = 0; nt < 4; ++nt)
        acc[mt][nt] = wmma_bf16(af[mt], bf[nt], acc[mt][nt]);
  }

  const int gn0   = n0 + wn * 64;
  const int mbase = m0 + wm * 64;
#pragma unroll
  for (int mt = 0; mt < 4; ++mt)
#pragma unroll
    for (int nt = 0; nt < 4; ++nt)
#pragma unroll
      for (int r = 0; r < 8; ++r) {
        const int n = gn0 + nt * 16 + ln;
        const int s = mbase + mt * 16 + hf * 8 + r;
        if (n < kH) out[(size_t)s * kH + n] = acc[mt][nt][r];
      }
}

// ---------------------------------------------------------------- launcher
extern "C" void kernel_launch(void* const* d_in, const int* in_sizes, int n_in,
                              void* d_out, int out_size, void* d_ws,
                              size_t ws_size, hipStream_t stream) {
  const int*   positions = (const int*)d_in[0];
  const float* hidden    = (const float*)d_in[1];
  const float* qkv_w     = (const float*)d_in[2];
  const float* o_w       = (const float*)d_in[3];
  const float* sinks     = (const float*)d_in[4];
  float* out = (float*)d_out;

  char* p = (char*)d_ws;
  auto alloc = [&](size_t bytes) -> char* {
    char* r = p;
    p += (bytes + 255) & ~(size_t)255;
    return r;
  };
  __bf16* hsb  = (__bf16*)alloc((size_t)kS * kH * 2);
  __bf16* wqb  = (__bf16*)alloc((size_t)kQKV * kH * 2);
  __bf16* wob  = (__bf16*)alloc((size_t)kH * kQD * 2);
  __bf16* qbuf = (__bf16*)alloc((size_t)kS * kQD * 2);
  __bf16* kbuf = (__bf16*)alloc((size_t)kS * kKVD * 2);
  __bf16* vtb  = (__bf16*)alloc((size_t)kNKV * kHD * kS * 2);
  __bf16* attn = (__bf16*)alloc((size_t)kS * kQD * 2);

  {
    int n = kS * kH;
    cvt_kernel<<<(n / 8 + 255) / 256, 256, 0, stream>>>(hidden, hsb, n);
  }
  {
    int n = kQKV * kH;
    cvt_kernel<<<(n / 8 + 255) / 256, 256, 0, stream>>>(qkv_w, wqb, n);
  }
  {
    int n = kH * kQD;
    cvt_kernel<<<(n / 8 + 255) / 256, 256, 0, stream>>>(o_w, wob, n);
  }

  qkv_gemm_rope_kernel<<<dim3(kQKV / 256, kS / 128), 256, 0, stream>>>(
      hsb, wqb, positions, qbuf, kbuf, vtb);
  attn_kernel<<<kNH * (kS / 16), 32, 0, stream>>>(qbuf, kbuf, vtb, sinks,
                                                  attn);
  o_gemm_kernel<<<dim3((kH + 255) / 256, kS / 128), 256, 0, stream>>>(
      attn, wob, out);
}
